// MultiHeadSelfAttention_60524679135759
// MI455X (gfx1250) — matrix operand order repaired and checked by reading, not compiled, not run on hardware
//
#include <hip/hip_runtime.h>

typedef _Float16 v16h __attribute__((ext_vector_type(16)));
typedef _Float16 v8h  __attribute__((ext_vector_type(8)));
typedef float    v8f  __attribute__((ext_vector_type(8)));
typedef int      v4i  __attribute__((ext_vector_type(4)));

#define DIM   1024
#define HEADS 16
#define HD    64
#define BSZ   2
#define TLEN  2048
#define MTOT  (BSZ*TLEN)   /* 4096 */
#define N3    (3*DIM)      /* 3072 */

#define LDS_STRIDE 40      /* 32 halfs + 8 pad -> 80B row stride, conflict-free */

#if __has_builtin(__builtin_amdgcn_global_load_async_to_lds_b128)
#define HAS_ASYNC_LDS 1
#else
#define HAS_ASYNC_LDS 0
#endif

static __device__ __forceinline__ v8f wmma16(v16h a, v16h b, v8f c) {
  return __builtin_amdgcn_wmma_f32_16x16x32_f16(false, a, false, b, (short)0, c,
                                                false, false);
}

// 16-byte global -> LDS copy; async (ASYNCcnt) when the builtin exists.
static __device__ __forceinline__ void copy16_g2l(void* lds, const void* g) {
#if HAS_ASYNC_LDS
  __builtin_amdgcn_global_load_async_to_lds_b128(
      (__attribute__((address_space(1))) v4i*)g,
      (__attribute__((address_space(3))) v4i*)lds, 0, 0);
#else
  *(v8h*)lds = *(const v8h*)g;
#endif
}

static __device__ __forceinline__ void async_wait0() {
#if HAS_ASYNC_LDS
#if __has_builtin(__builtin_amdgcn_s_wait_asynccnt)
  __builtin_amdgcn_s_wait_asynccnt(0);
#else
  asm volatile("s_wait_asynccnt 0" ::: "memory");
#endif
#endif
}

// A fragment (16x32 f16) from row-major [*, ldk] at (row m, col k).
// contiguous k order (this code's choice, matches load_b): lane l: m = l&15, g = l>>4; a[i] = A[m][ 16g+i ]
static __device__ __forceinline__ v16h load_a(const _Float16* base, int ldk,
                                              int m, int k, int lane) {
  int g = lane >> 4;
  const _Float16* p = base + (long)m * ldk + k;
  v8h lo = *(const v8h*)(p + 16 * g);
  v8h hi = *(const v8h*)(p + 16 * g + 8);
  v16h a;
#pragma unroll
  for (int i = 0; i < 8; i++) { a[i] = lo[i]; a[i + 8] = hi[i]; }
  return a;
}

// B fragment (32x16 f16), column n, K-major storage (Bt[n][k] contiguous in k).
// lane l: n = l&15, g = l>>4; b[i] = B[k + 16g + i][n]
static __device__ __forceinline__ v16h load_b(const _Float16* bt, int ldk,
                                              int n, int k, int lane) {
  int g = lane >> 4;
  return *(const v16h*)(bt + (long)n * ldk + k + 16 * g);
}

__global__ void f32_to_f16_kernel(const float* __restrict__ in,
                                  _Float16* __restrict__ out, int n) {
  for (int i = blockIdx.x * blockDim.x + threadIdx.x; i < n;
       i += gridDim.x * blockDim.x)
    out[i] = (_Float16)in[i];
}

// Wt[n][k] = (f16) W[k][n]   (W is [K][N] row-major f32)
__global__ void transpose_f16_kernel(const float* __restrict__ W,
                                     _Float16* __restrict__ Wt, int K, int N) {
  int i = blockIdx.x * blockDim.x + threadIdx.x;
  if (i >= K * N) return;
  int k = i / N, n = i - k * N;
  Wt[(long)n * K + k] = (_Float16)W[(long)k * N + n];
}

// LDS double-buffered WMMA GEMM: 256 threads (8 waves), 128x128 tile, k-step 32.
// Wave (wm,wn) computes a 64(M) x 32(N) sub-tile = 4x2 WMMA accumulators.
__global__ __launch_bounds__(256) void gemm_lds_kernel(
    const _Float16* __restrict__ A, const _Float16* __restrict__ Bt,
    const float* __restrict__ bias, float* __restrict__ Cf,
    _Float16* __restrict__ Ch, int M, int N, int K) {
  __shared__ _Float16 ldsA[2][128][LDS_STRIDE];
  __shared__ _Float16 ldsB[2][128][LDS_STRIDE];
  int tid = threadIdx.x;
  int lane = tid & 31;
  int ln = lane & 15, g = lane >> 4;
  int wave = tid >> 5;
  int wm = wave >> 2, wn = wave & 3;
  long tileM = (long)blockIdx.y * 128;
  long tileN = (long)blockIdx.x * 128;

  // stage one 128x32 A tile + 128x32 B tile: 512 16B chunks each -> 2 per thread
  auto issue_copy = [&](int buf, int k) {
#pragma unroll
    for (int it = 0; it < 2; it++) {
      int c = tid + 256 * it;
      int row = c >> 2, seg = (c & 3) * 8;
      copy16_g2l(&ldsA[buf][row][seg], A + (tileM + row) * K + k + seg);
      copy16_g2l(&ldsB[buf][row][seg], Bt + (tileN + row) * K + k + seg);
      if (k + 64 < K) {  // cover latency 2 stages deep (global_prefetch_b8)
        __builtin_prefetch(A + (tileM + row) * K + k + 64 + seg, 0, 0);
        __builtin_prefetch(Bt + (tileN + row) * K + k + 64 + seg, 0, 0);
      }
    }
  };

  v8f acc[4][2] = {};
  int nk = K / 32;
  issue_copy(0, 0);
  for (int kt = 0; kt < nk; kt++) {
    int buf = kt & 1;
    async_wait0();      // own async copies into `buf` done
    __syncthreads();    // everyone's copies visible
    if (kt + 1 < nk) issue_copy(buf ^ 1, 32 * (kt + 1));

    v16h bfrag[2];
#pragma unroll
    for (int j = 0; j < 2; j++) {
      int n = wn * 32 + 16 * j + ln;
      v8h blo = *(const v8h*)&ldsB[buf][n][8 * g];
      v8h bhi = *(const v8h*)&ldsB[buf][n][16 + 8 * g];
#pragma unroll
      for (int q = 0; q < 8; q++) { bfrag[j][q] = blo[q]; bfrag[j][q + 8] = bhi[q]; }
    }
#pragma unroll
    for (int i = 0; i < 4; i++) {
      int m = wm * 64 + 16 * i + ln;
      v8h alo = *(const v8h*)&ldsA[buf][m][8 * g];
      v8h ahi = *(const v8h*)&ldsA[buf][m][16 + 8 * g];
      v16h a;
#pragma unroll
      for (int q = 0; q < 8; q++) { a[q] = alo[q]; a[q + 8] = ahi[q]; }
#pragma unroll
      for (int j = 0; j < 2; j++) acc[i][j] = wmma16(a, bfrag[j], acc[i][j]);
    }
    __syncthreads();    // all reads of `buf` done before it is refilled
  }

#pragma unroll
  for (int i = 0; i < 4; i++) {
#pragma unroll
    for (int j = 0; j < 2; j++) {
      long n = tileN + wn * 32 + 16 * j + ln;
      float bs = bias[n];
#pragma unroll
      for (int r = 0; r < 8; r++) {
        long row = tileM + wm * 64 + 16 * i + r + 8 * g;
        float v = acc[i][j][r] + bs;
        if (Cf) Cf[row * N + n] = v;
        else    Ch[row * N + n] = (_Float16)v;
      }
    }
  }
}

// RoPE + head split. qkv f16 [B*T][3072] -> Qh/Kh [bh][t][64] (Q pre-scaled by
// 1/sqrt(64)), Vt [bh][d][T]. One thread per (bt, h, dpair<32).
__global__ void rope_split_kernel(const _Float16* __restrict__ qkv,
                                  const int* __restrict__ posp,
                                  _Float16* __restrict__ Qh,
                                  _Float16* __restrict__ Kh,
                                  _Float16* __restrict__ Vt) {
  int idx = blockIdx.x * blockDim.x + threadIdx.x;
  if (idx >= MTOT * HEADS * 32) return;
  int dp = idx & 31;
  int h = (idx >> 5) & (HEADS - 1);
  int bt = idx >> 9;
  int b = bt / TLEN, t = bt - b * TLEN;
  int pos = posp[0] + t;
  float theta = __powf(10000.0f, -(float)dp * (1.0f / 32.0f));
  float ang = (float)pos * theta;
  float s = __sinf(ang), c = __cosf(ang);
  const _Float16* base = qkv + (long)bt * N3 + h * HD;
  float q1 = (float)base[2 * dp],           q2 = (float)base[2 * dp + 1];
  float k1 = (float)base[DIM + 2 * dp],     k2 = (float)base[DIM + 2 * dp + 1];
  float v1 = (float)base[2 * DIM + 2 * dp], v2 = (float)base[2 * DIM + 2 * dp + 1];
  const float scale = 0.125f;  // 1/sqrt(HD), folded into Q
  long bh = (long)b * HEADS + h;
  _Float16* qrow = Qh + (bh * TLEN + t) * HD;
  _Float16* krow = Kh + (bh * TLEN + t) * HD;
  qrow[dp]      = (_Float16)((q1 * c - q2 * s) * scale);
  qrow[dp + 32] = (_Float16)((q1 * s + q2 * c) * scale);
  krow[dp]      = (_Float16)(k1 * c - k2 * s);
  krow[dp + 32] = (_Float16)(k1 * s + k2 * c);
  Vt[(bh * HD + 2 * dp) * TLEN + t]     = (_Float16)v1;
  Vt[(bh * HD + 2 * dp + 1) * TLEN + t] = (_Float16)v2;
}

// Flash attention: grid (T/64, B*HEADS), 4 waves; wave owns 16 query rows.
__global__ __launch_bounds__(128) void attn_kernel(
    const _Float16* __restrict__ Qh, const _Float16* __restrict__ Kh,
    const _Float16* __restrict__ Vt, _Float16* __restrict__ Oh) {
  __shared__ _Float16 ldsP[4][16][32];
  int lane = threadIdx.x & 31;
  int wave = threadIdx.x >> 5;
  int ln = lane & 15, g = lane >> 4;
  int bh = blockIdx.y;
  int q0 = blockIdx.x * 64 + wave * 16;
  const _Float16* Qb = Qh + (long)bh * TLEN * HD;
  const _Float16* Kb = Kh + (long)bh * TLEN * HD;
  const _Float16* Vb = Vt + (long)bh * HD * TLEN;

  v16h qa[2];
#pragma unroll
  for (int s = 0; s < 2; s++) qa[s] = load_a(Qb, HD, q0 + ln, 32 * s, lane);

  v8f Of[4] = {};
  float mrow[8], lrow[8];
#pragma unroll
  for (int r = 0; r < 8; r++) { mrow[r] = -3.0e38f; lrow[r] = 0.f; }

  int kEnd = blockIdx.x * 64 + 64;  // causal truncation for this query tile
  for (int kb = 0; kb < kEnd; kb += 32) {
    v8f s0 = {}, s1 = {};
    {
      v16h b00 = load_b(Kb, HD, kb + ln, 0, lane);
      v16h b01 = load_b(Kb, HD, kb + ln, 32, lane);
      s0 = wmma16(qa[0], b00, s0);
      s0 = wmma16(qa[1], b01, s0);
      v16h b10 = load_b(Kb, HD, kb + 16 + ln, 0, lane);
      v16h b11 = load_b(Kb, HD, kb + 16 + ln, 32, lane);
      s1 = wmma16(qa[0], b10, s1);
      s1 = wmma16(qa[1], b11, s1);
    }
#pragma unroll
    for (int r = 0; r < 8; r++) {
      int q = q0 + r + 8 * g;
      if (kb + ln > q)      s0[r] = -3.0e38f;
      if (kb + 16 + ln > q) s1[r] = -3.0e38f;
    }
    float p0[8], p1[8];
#pragma unroll
    for (int r = 0; r < 8; r++) {
      float mx = fmaxf(s0[r], s1[r]);
#pragma unroll
      for (int d = 1; d < 16; d <<= 1) mx = fmaxf(mx, __shfl_xor(mx, d, 32));
      float mnew = fmaxf(mrow[r], mx);
      float corr = __expf(mrow[r] - mnew);
      mrow[r] = mnew;
      float e0 = __expf(s0[r] - mnew);
      float e1 = __expf(s1[r] - mnew);
      p0[r] = e0; p1[r] = e1;
      float rs = e0 + e1;
#pragma unroll
      for (int d = 1; d < 16; d <<= 1) rs += __shfl_xor(rs, d, 32);
      lrow[r] = lrow[r] * corr + rs;
#pragma unroll
      for (int j = 0; j < 4; j++) Of[j][r] *= corr;
    }
#pragma unroll
    for (int r = 0; r < 8; r++) {
      ldsP[wave][r + 8 * g][ln]      = (_Float16)p0[r];
      ldsP[wave][r + 8 * g][16 + ln] = (_Float16)p1[r];
    }
    __syncthreads();
    v16h pa = load_a(&ldsP[wave][0][0], 32, ln, 0, lane);
#pragma unroll
    for (int j = 0; j < 4; j++) {
      v16h bv = load_b(Vb, TLEN, 16 * j + ln, kb, lane);
      Of[j] = wmma16(pa, bv, Of[j]);
    }
    __syncthreads();
  }

  int b = bh >> 4, h = bh & 15;
#pragma unroll
  for (int j = 0; j < 4; j++) {
    int d = 16 * j + ln;
#pragma unroll
    for (int r = 0; r < 8; r++) {
      int q = q0 + r + 8 * g;
      float o = Of[j][r] / lrow[r];
      Oh[((long)(b * TLEN + q)) * DIM + h * HD + d] = (_Float16)o;
    }
  }
}

extern "C" void kernel_launch(void* const* d_in, const int* in_sizes, int n_in,
                              void* d_out, int out_size, void* d_ws,
                              size_t ws_size, hipStream_t stream) {
  const float* x     = (const float*)d_in[0];
  const float* Wqkv  = (const float*)d_in[1];
  const float* bqkv  = (const float*)d_in[2];
  const float* Wproj = (const float*)d_in[3];
  const float* bproj = (const float*)d_in[4];
  const int*   pos   = (const int*)d_in[5];
  float* out = (float*)d_out;

  char* ws = (char*)d_ws;
  size_t off = 0;
  auto alloc = [&](size_t bytes) -> char* {
    char* p = ws + off;
    off += (bytes + 255) & ~(size_t)255;
    return p;
  };
  _Float16* xh     = (_Float16*)alloc((size_t)MTOT * DIM * 2);
  _Float16* wqkvT  = (_Float16*)alloc((size_t)N3 * DIM * 2);
  _Float16* wprojT = (_Float16*)alloc((size_t)DIM * DIM * 2);
  _Float16* qkvh   = (_Float16*)alloc((size_t)MTOT * N3 * 2);
  _Float16* Qh     = (_Float16*)alloc((size_t)BSZ * HEADS * TLEN * HD * 2);
  _Float16* Kh     = (_Float16*)alloc((size_t)BSZ * HEADS * TLEN * HD * 2);
  _Float16* Vt     = (_Float16*)alloc((size_t)BSZ * HEADS * TLEN * HD * 2);
  _Float16* attnh  = (_Float16*)alloc((size_t)MTOT * DIM * 2);
  (void)ws_size; (void)in_sizes; (void)n_in; (void)out_size;

  f32_to_f16_kernel<<<1024, 256, 0, stream>>>(x, xh, MTOT * DIM);
  transpose_f16_kernel<<<(DIM * N3 + 255) / 256, 256, 0, stream>>>(Wqkv, wqkvT, DIM, N3);
  transpose_f16_kernel<<<(DIM * DIM + 255) / 256, 256, 0, stream>>>(Wproj, wprojT, DIM, DIM);

  dim3 g1(N3 / 128, MTOT / 128);  // 24 x 32
  gemm_lds_kernel<<<g1, 256, 0, stream>>>(xh, wqkvT, bqkv, nullptr, qkvh,
                                          MTOT, N3, DIM);

  int nr = MTOT * HEADS * 32;
  rope_split_kernel<<<(nr + 255) / 256, 256, 0, stream>>>(qkvh, pos, Qh, Kh, Vt);

  dim3 g2(TLEN / 64, BSZ * HEADS);
  attn_kernel<<<g2, 128, 0, stream>>>(Qh, Kh, Vt, attnh);

  dim3 g3(DIM / 128, MTOT / 128);  // 8 x 32
  gemm_lds_kernel<<<g3, 256, 0, stream>>>(attnh, wprojT, bproj, out, nullptr,
                                          MTOT, DIM, DIM);
}
